// StableHyperNet_15522011808520
// MI455X (gfx1250) — compile-verified, matrix operand order checked by reading, not run on hardware
//
#include <hip/hip_runtime.h>
#include <hip/hip_bf16.h>
#include <cmath>

// ---------------------------------------------------------------------------
// Problem constants
// ---------------------------------------------------------------------------
#define Dd     512
#define Cc     100
#define Tt     15
#define Hh     5
#define ENCd   64
#define GENd   128
#define NINT   15
#define NLEAF  16
#define TOTALp 139440           // T*N_INT*(D+1) + T*N_LEAF*C + T
#define NSUP   4096
#define NQRY   32768
#define I0p    115200           // T*N_INT*D
#define I1p    115425           // + T*N_INT
#define I2p    139425           // + T*N_LEAF*C
#define JROWS  1200             // (H*T)*16  (n padded 15 -> 16)
#define KPAD   1216             // JROWS padded to multiple of 32 for WMMA K
#define JT_N   75               // JROWS / 16 column tiles

// ---------------------------------------------------------------------------
// Workspace layout (bytes, 256-aligned).  Total ~6.5 MB.
// [0, ZSPAN) is zero-initialized every launch (G accum, padded splitW/B, WoutT)
// ---------------------------------------------------------------------------
#define OFF_G      0u           //  320 f32  : sum_s gelu(ln(h1))          (zeroed)
#define OFF_SPLITW 1536u        // 1200x512 bf16, zero pad rows n==15      (zeroed)
#define OFF_SPLITB 1230336u     // 1200 f32                               (zeroed)
#define OFF_WOUTT  1235200u     // 112x1216 bf16 transposed out-weights    (zeroed)
#define ZSPAN      1507584u
#define OFF_LEAF   1507584u     // 1200x100 f32 raw leaf logits
#define OFF_TREE   1987584u     // 75 f32 raw tree scores
#define OFF_TW     1988096u     // 75 f32 softmax(tree)*softmax(hw)
#define OFF_GBUF   1988608u     // 640 f32 generator hidden g[h][128]
#define OFF_XS     1991168u     // 4096x512 bf16 X_support
#define OFF_W1T    6185472u     // 5x64x512 bf16 enc_w1 transposed (e-major)
// end ~6513152 bytes

typedef __bf16 v16bf __attribute__((ext_vector_type(16)));
typedef float  v8f   __attribute__((ext_vector_type(8)));

union FragU { v16bf v; uint4 u[2]; };

// A/B fragment loader for v_wmma_f32_16x16x32_bf16 (wave32):
// caller passes p = base + (row16 + (lane&15))*ld + k + 8*(lane>>4);
// fragment = 8 bf16 at p (K k..k+7 half) and 8 bf16 at p+16 (K k+16.. half).
__device__ __forceinline__ v16bf load_frag(const __hip_bfloat16* p) {
  FragU f;
  f.u[0] = *reinterpret_cast<const uint4*>(p);
  f.u[1] = *reinterpret_cast<const uint4*>(p + 16);
  return f.v;
}

__device__ __forceinline__ float gelu_exact(float x) {
  return 0.5f * x * (1.0f + erff(x * 0.70710678118654752f));
}

// ---------------------------------------------------------------------------
__global__ void k_zero(uint32_t* p, int n) {
  int i = blockIdx.x * blockDim.x + threadIdx.x;
  if (i < n) p[i] = 0u;
}

__global__ void k_f32_to_bf16(const float* __restrict__ src,
                              __hip_bfloat16* __restrict__ dst, int n) {
  int i = blockIdx.x * blockDim.x + threadIdx.x;
  if (i < n) dst[i] = __float2bfloat16(src[i]);
}

// enc_w1 (H,D,ENC) -> w1t (H,ENC,D) bf16 so B-fragments read K=d contiguously
__global__ void k_w1t(const float* __restrict__ w1, __hip_bfloat16* __restrict__ dst) {
  int i = blockIdx.x * blockDim.x + threadIdx.x;
  if (i >= Hh * ENCd * Dd) return;
  int h = i / (ENCd * Dd);
  int r = i - h * ENCd * Dd;
  int e = r / Dd, d = r - e * Dd;
  dst[i] = __float2bfloat16(w1[(h * Dd + d) * ENCd + e]);
}

// ---------------------------------------------------------------------------
// Encoder: per (head, 64-support tile): WMMA (64x512)@(512x64) -> h1 in LDS,
// layernorm + gelu per row, column-sum -> atomic accumulate into G[h][e].
// ---------------------------------------------------------------------------
__global__ void __launch_bounds__(256)
k_encoder(const __hip_bfloat16* __restrict__ Xs, const __hip_bfloat16* __restrict__ w1t,
          const float* __restrict__ b1, const float* __restrict__ lng,
          const float* __restrict__ lnb, float* __restrict__ G) {
  __shared__ float h1s[64][64];
  int h  = blockIdx.x >> 6;
  int s0 = (blockIdx.x & 63) << 6;
  int tid  = threadIdx.x;
  int wave = tid >> 5, lane = tid & 31;
  int r16  = lane & 15, half = lane >> 4;
  int rt   = wave >> 1;                      // row tile (0..3)
  const __hip_bfloat16* w1h = w1t + (size_t)h * ENCd * Dd;

  // two col tiles per wave, shared A fragment -> two independent wmma chains
  {
    int ct0 = ((wave & 1) << 1);
    int ct1 = ct0 | 1;
    v8f acc0 = {}, acc1 = {};
    const __hip_bfloat16* arow = Xs  + (size_t)(s0 + rt * 16 + r16) * Dd + half * 8;
    const __hip_bfloat16* br0  = w1h + (size_t)(ct0 * 16 + r16) * Dd + half * 8;
    const __hip_bfloat16* br1  = w1h + (size_t)(ct1 * 16 + r16) * Dd + half * 8;
    for (int k = 0; k < Dd; k += 32) {
      v16bf a  = load_frag(arow + k);
      v16bf b0 = load_frag(br0 + k);
      v16bf b1f = load_frag(br1 + k);
      acc0 = __builtin_amdgcn_wmma_f32_16x16x32_bf16(false, a, false, b0,
                                                     (short)0, acc0, false, false);
      acc1 = __builtin_amdgcn_wmma_f32_16x16x32_bf16(false, a, false, b1f,
                                                     (short)0, acc1, false, false);
    }
    float bv0 = b1[h * ENCd + ct0 * 16 + r16];
    float bv1 = b1[h * ENCd + ct1 * 16 + r16];
    for (int i = 0; i < 8; ++i) {
      h1s[rt * 16 + 8 * half + i][ct0 * 16 + r16] = acc0[i] + bv0;
      h1s[rt * 16 + 8 * half + i][ct1 * 16 + r16] = acc1[i] + bv1;
    }
  }
  __syncthreads();

  if (tid < 64) {                            // layernorm + gelu, one row each
    int r = tid;
    float mu = 0.f;
    for (int e = 0; e < 64; ++e) mu += h1s[r][e];
    mu *= (1.f / 64.f);
    float var = 0.f;
    for (int e = 0; e < 64; ++e) { float d = h1s[r][e] - mu; var += d * d; }
    var *= (1.f / 64.f);
    float rs = rsqrtf(var + 1e-5f);
    for (int e = 0; e < 64; ++e) {
      float v = (h1s[r][e] - mu) * rs * lng[h * ENCd + e] + lnb[h * ENCd + e];
      h1s[r][e] = gelu_exact(v);
    }
  }
  __syncthreads();
  if (tid < 64) {                            // column partial sums
    float s = 0.f;
    for (int r = 0; r < 64; ++r) s += h1s[r][tid];
    atomicAdd(&G[h * ENCd + tid], s);
  }
}

// ---------------------------------------------------------------------------
// ctx = (G/NSUP) @ enc_w2 + b2 ;  g = gelu(ctx @ gen_w1 + b1)   (tiny)
// ---------------------------------------------------------------------------
__global__ void __launch_bounds__(256)
k_ctx(const float* __restrict__ G, const float* __restrict__ w2,
      const float* __restrict__ b2, const float* __restrict__ gw1,
      const float* __restrict__ gb1, float* __restrict__ gbuf) {
  __shared__ float gm[Hh][ENCd];
  __shared__ float ctx[Hh][ENCd];
  int tid = threadIdx.x;
  for (int i = tid; i < Hh * ENCd; i += 256) gm[i >> 6][i & 63] = G[i] * (1.f / NSUP);
  __syncthreads();
  for (int i = tid; i < Hh * ENCd; i += 256) {
    int h = i >> 6, f = i & 63;
    float s = b2[i];
    for (int e = 0; e < ENCd; ++e) s += gm[h][e] * w2[(h * ENCd + e) * ENCd + f];
    ctx[h][f] = s;
  }
  __syncthreads();
  for (int i = tid; i < Hh * GENd; i += 256) {
    int h = i >> 7, j = i & 127;
    float s = gb1[i];
    for (int e = 0; e < ENCd; ++e) s += ctx[h][e] * gw1[(h * ENCd + e) * GENd + j];
    gbuf[i] = gelu_exact(s);
  }
}

// ---------------------------------------------------------------------------
// params GEMV: streams 357 MB of gen_w2 (bandwidth bound; prefetch ahead),
// tanh*2, and scatters into split_w(bf16, n padded) / split_b / leaf / tree.
// ---------------------------------------------------------------------------
__global__ void __launch_bounds__(256)
k_params(const float* __restrict__ gbuf, const float* __restrict__ gw2,
         const float* __restrict__ gb2, __hip_bfloat16* __restrict__ splitW,
         float* __restrict__ splitB, float* __restrict__ leafRaw,
         float* __restrict__ treeRaw) {
  int tid = blockIdx.x * 256 + threadIdx.x;
  if (tid >= Hh * TOTALp) return;
  int h = tid / TOTALp;
  int p = tid - h * TOTALp;
  const float* col = gw2 + (size_t)h * GENd * TOTALp + p;
  float acc = 0.f;
  for (int k = 0; k < GENd; ++k) {
    if (k + 8 < GENd) __builtin_prefetch(col + (size_t)(k + 8) * TOTALp, 0, 0);
    acc += gbuf[h * GENd + k] * col[(size_t)k * TOTALp];
  }
  float v = 2.f * tanhf(acc + gb2[(size_t)h * TOTALp + p]);

  if (p < I0p) {
    int t = p / (NINT * Dd);
    int rm = p - t * NINT * Dd;
    int n = rm >> 9, d = rm & 511;
    splitW[((size_t)((h * Tt + t) * 16 + n) << 9) + d] = __float2bfloat16(v);
  } else if (p < I1p) {
    int q = p - I0p; int t = q / NINT, n = q - t * NINT;
    splitB[(h * Tt + t) * 16 + n] = v;
  } else if (p < I2p) {
    int r = p - I1p;
    int t = r / (NLEAF * Cc);
    int rr = r - t * NLEAF * Cc;
    int l = rr / Cc, c = rr - l * Cc;
    leafRaw[(size_t)((h * Tt + t) * 16 + l) * Cc + c] = v;
  } else {
    treeRaw[h * Tt + (p - I2p)] = v;
  }
}

// softmax(head_weights) and per-head softmax(tree scores) folded: tw = sm_T*sm_H
__global__ void k_tw(const float* __restrict__ hwraw, const float* __restrict__ treeRaw,
                     float* __restrict__ tw) {
  __shared__ float hwsm[Hh];
  int tid = threadIdx.x;
  if (tid == 0) {
    float mx = hwraw[0];
    for (int h = 1; h < Hh; ++h) mx = fmaxf(mx, hwraw[h]);
    float e[Hh], s = 0.f;
    for (int h = 0; h < Hh; ++h) { e[h] = expf(hwraw[h] - mx); s += e[h]; }
    for (int h = 0; h < Hh; ++h) hwsm[h] = e[h] / s;
  }
  __syncthreads();
  if (tid < Hh) {
    int h = tid;
    float mx = treeRaw[h * Tt];
    for (int t = 1; t < Tt; ++t) mx = fmaxf(mx, treeRaw[h * Tt + t]);
    float s = 0.f;
    for (int t = 0; t < Tt; ++t) s += expf(treeRaw[h * Tt + t] - mx);
    for (int t = 0; t < Tt; ++t)
      tw[h * Tt + t] = expf(treeRaw[h * Tt + t] - mx) / s * hwsm[h];
  }
}

// leaf softmax over C=100, scaled by tw, written TRANSPOSED: WoutT[c][jl] (bf16)
__global__ void __launch_bounds__(128)
k_leaf(const float* __restrict__ leafRaw, const float* __restrict__ tw,
       const float* __restrict__ temperature, __hip_bfloat16* __restrict__ WoutT) {
  __shared__ float red[128];
  int row = blockIdx.x, tid = threadIdx.x;
  int h = row / (Tt * 16);
  int t = (row - h * Tt * 16) >> 4;
  float temp = fminf(fmaxf(temperature[0], 0.1f), 2.0f);
  float x = (tid < Cc) ? leafRaw[(size_t)row * Cc + tid] / temp : -3.0e38f;
  red[tid] = x; __syncthreads();
  for (int s = 64; s > 0; s >>= 1) { if (tid < s) red[tid] = fmaxf(red[tid], red[tid + s]); __syncthreads(); }
  float mx = red[0]; __syncthreads();
  float e = (tid < Cc) ? expf(x - mx) : 0.f;
  red[tid] = e; __syncthreads();
  for (int s = 64; s > 0; s >>= 1) { if (tid < s) red[tid] += red[tid + s]; __syncthreads(); }
  float inv = 1.f / red[0];
  if (tid < Cc) WoutT[(size_t)tid * KPAD + row] = __float2bfloat16(e * inv * tw[h * Tt + t]);
}

// ---------------------------------------------------------------------------
// Fused main kernel, 16 queries per workgroup (8 waves):
//  stage 0: convert this block's 16x512 f32 query tile -> bf16 in LDS
//           (X_query is partitioned across blocks: no global bf16 round-trip)
//  phase 1: WMMA (16xD)@(Dx1200), jt/jt+8 paired on one A fragment
//           -> sigmoid -> depth-4 tree routing -> reach tile in LDS
//  phase 2: WMMA (16x1216 LDS)@(1216x112), dual K-chains -> out[q][c] f32
// LDS: 38912 + 8704 + 16384 = 64000 B
// ---------------------------------------------------------------------------
__global__ void __launch_bounds__(256)
k_main(const float* __restrict__ Xq, const __hip_bfloat16* __restrict__ splitW,
       const float* __restrict__ splitB, const __hip_bfloat16* __restrict__ WoutT,
       const float* __restrict__ temperature, float* __restrict__ out) {
  __shared__ __align__(16) __hip_bfloat16 Rbuf[16][KPAD];   // 38.9 KB
  __shared__ __align__(16) __hip_bfloat16 Atile[16][Dd];    // 16.0 KB
  __shared__ float decT[8][16][17];                          //  8.7 KB (per wave)
  int tid  = threadIdx.x;
  int wave = tid >> 5, lane = tid & 31;
  int r16  = lane & 15, half = lane >> 4;
  int q0   = blockIdx.x << 4;
  float itemp = 1.f / fminf(fmaxf(temperature[0], 0.1f), 2.0f);

  // stage 0: A tile f32->bf16 into LDS (coalesced), plus zero K-pad columns
  for (int i = tid; i < 16 * Dd; i += 256) {
    int r = i >> 9, d = i & 511;
    Atile[r][d] = __float2bfloat16(Xq[(size_t)(q0 + r) * Dd + d]);
  }
  Rbuf[tid >> 4][JROWS + (tid & 15)] = __float2bfloat16(0.f);
  __syncthreads();

  // ---- phase 1 ----  (all loop bounds wave-uniform -> EXEC all-1s at wmma)
  const __hip_bfloat16* arow = &Atile[r16][half * 8];
  for (int jt = wave; jt < JT_N; jt += 16) {
    int jt2 = jt + 8;
    bool has2 = (jt2 < JT_N);                       // wave-uniform
    int jt2c = has2 ? jt2 : jt;                     // keep pointer in-bounds
    v8f acc0 = {}, acc1 = {};
    const __hip_bfloat16* br0 = splitW + (size_t)(jt  * 16 + r16) * Dd + half * 8;
    const __hip_bfloat16* br1 = splitW + (size_t)(jt2c * 16 + r16) * Dd + half * 8;
    for (int k = 0; k < Dd; k += 32) {
      v16bf a  = load_frag(arow + k);
      v16bf b0 = load_frag(br0 + k);
      acc0 = __builtin_amdgcn_wmma_f32_16x16x32_bf16(false, a, false, b0,
                                                     (short)0, acc0, false, false);
      if (has2) {
        v16bf b1f = load_frag(br1 + k);
        acc1 = __builtin_amdgcn_wmma_f32_16x16x32_bf16(false, a, false, b1f,
                                                       (short)0, acc1, false, false);
      }
    }
    for (int pass = 0; pass < 2; ++pass) {
      if (pass == 1 && !has2) break;
      int jtc = pass ? jt2 : jt;
      v8f acc = pass ? acc1 : acc0;
      // sigmoid epilogue: this lane owns column n=r16, rows i+8*half
      float bv = splitB[jtc * 16 + r16];
      for (int i = 0; i < 8; ++i) {
        float x = (acc[i] + bv) * itemp;
        decT[wave][8 * half + i][r16] = 1.f / (1.f + expf(-x));
      }
      // tree routing (intra-wave LDS ops in order): lane = row r16,
      // leaves half*8..+7; ancestor at depth d: (2^d-1) + (l>>(4-d))
      for (int li = 0; li < 8; ++li) {
        int l = half * 8 + li;
        float pr = 1.f;
        for (int dl = 0; dl < 4; ++dl) {
          int node = (1 << dl) - 1 + (l >> (4 - dl));
          float dv = decT[wave][r16][node];
          pr *= ((l >> (3 - dl)) & 1) ? dv : (1.f - dv);
        }
        Rbuf[r16][jtc * 16 + l] = __float2bfloat16(pr);
      }
    }
  }
  __syncthreads();

  // ---- phase 2 ---- (wave-uniform skip keeps EXEC all-1s inside WMMA)
  if (wave < 7) {
    int ct = wave;                                  // 7 tiles cover c = 0..111
    v8f acc0 = {}, acc1 = {};
    const __hip_bfloat16* a2 = &Rbuf[r16][half * 8];
    const __hip_bfloat16* b2 = WoutT + (size_t)(ct * 16 + r16) * KPAD + half * 8;
    for (int k = 0; k < KPAD; k += 64) {            // 19 exact pairs
      acc0 = __builtin_amdgcn_wmma_f32_16x16x32_bf16(false, load_frag(a2 + k),
                                                     false, load_frag(b2 + k),
                                                     (short)0, acc0, false, false);
      acc1 = __builtin_amdgcn_wmma_f32_16x16x32_bf16(false, load_frag(a2 + k + 32),
                                                     false, load_frag(b2 + k + 32),
                                                     (short)0, acc1, false, false);
    }
    v8f acc = acc0 + acc1;
    int c = ct * 16 + r16;
    if (c < Cc) {
      for (int i = 0; i < 8; ++i)
        out[(size_t)(q0 + 8 * half + i) * Cc + c] = acc[i];
    }
  }
}

// ---------------------------------------------------------------------------
extern "C" void kernel_launch(void* const* d_in, const int* in_sizes, int n_in,
                              void* d_out, int out_size, void* d_ws, size_t ws_size,
                              hipStream_t stream) {
  (void)in_sizes; (void)n_in; (void)out_size; (void)ws_size; // needs ws >= ~6.6 MB
  const float* Xs_f   = (const float*)d_in[0];
  const float* Xq_f   = (const float*)d_in[1];
  const float* enc_w1 = (const float*)d_in[2];
  const float* enc_b1 = (const float*)d_in[3];
  const float* ln_g   = (const float*)d_in[4];
  const float* ln_b   = (const float*)d_in[5];
  const float* enc_w2 = (const float*)d_in[6];
  const float* enc_b2 = (const float*)d_in[7];
  const float* gen_w1 = (const float*)d_in[8];
  const float* gen_b1 = (const float*)d_in[9];
  const float* gen_w2 = (const float*)d_in[10];
  const float* gen_b2 = (const float*)d_in[11];
  const float* hw     = (const float*)d_in[12];
  const float* temp   = (const float*)d_in[13];

  char* ws = (char*)d_ws;
  float*          G       = (float*)(ws + OFF_G);
  __hip_bfloat16* splitW  = (__hip_bfloat16*)(ws + OFF_SPLITW);
  float*          splitB  = (float*)(ws + OFF_SPLITB);
  __hip_bfloat16* WoutT   = (__hip_bfloat16*)(ws + OFF_WOUTT);
  float*          leafRaw = (float*)(ws + OFF_LEAF);
  float*          treeRaw = (float*)(ws + OFF_TREE);
  float*          tw      = (float*)(ws + OFF_TW);
  float*          gbuf    = (float*)(ws + OFF_GBUF);
  __hip_bfloat16* Xs      = (__hip_bfloat16*)(ws + OFF_XS);
  __hip_bfloat16* w1t     = (__hip_bfloat16*)(ws + OFF_W1T);
  float* outp = (float*)d_out;

  k_zero<<<(ZSPAN / 4 + 255) / 256, 256, 0, stream>>>((uint32_t*)ws, ZSPAN / 4);
  k_f32_to_bf16<<<(NSUP * Dd + 255) / 256, 256, 0, stream>>>(Xs_f, Xs, NSUP * Dd);
  k_w1t<<<(Hh * ENCd * Dd + 255) / 256, 256, 0, stream>>>(enc_w1, w1t);
  k_encoder<<<Hh * (NSUP / 64), 256, 0, stream>>>(Xs, w1t, enc_b1, ln_g, ln_b, G);
  k_ctx<<<1, 256, 0, stream>>>(G, enc_w2, enc_b2, gen_w1, gen_b1, gbuf);
  k_params<<<(Hh * TOTALp + 255) / 256, 256, 0, stream>>>(gbuf, gen_w2, gen_b2,
                                                          splitW, splitB, leafRaw, treeRaw);
  k_tw<<<1, 32, 0, stream>>>(hw, treeRaw, tw);
  k_leaf<<<JROWS, 128, 0, stream>>>(leafRaw, tw, temp, WoutT);
  k_main<<<NQRY / 16, 256, 0, stream>>>(Xq_f, splitW, splitB, WoutT, temp, outp);
}
